// GAMoEmotion_8770323218992
// MI455X (gfx1250) — matrix operand order repaired and checked by reading, not compiled, not run on hardware
//
#include <hip/hip_runtime.h>
#include <hip/hip_bf16.h>
#include <math.h>

typedef __attribute__((ext_vector_type(16))) _Float16 v16h;
typedef __attribute__((ext_vector_type(8)))  _Float16 v8h;
typedef __attribute__((ext_vector_type(8)))  float    v8f;

union Frag { v16h v; v8h h2[2]; };

#define NB    64
#define NC2   35
#define TIN   512
#define TT    256
#define NE    70
#define NED   32
#define NCL   3
#define NTOK  (NB * NC2)   // 2240

// ---------------------------------------------------------------------------
// Register-blocked WMMA GEMM: C[M,N] = A[M,K] @ Bt[N,K]^T + bias.
// One wave computes a 32x32 tile (2x2 WMMA grid): each A/B fragment feeds two
// v_wmma_f32_16x16x32_f16, doubling wmma density per byte loaded.
// Requires M%32==0, N%32==0, K%32==0 (true for all call sites here).
// Fragment layouts follow CDNA5 ISA 7.12.2 (16-bit A 16x32, B 32x16, f32 C).
// ---------------------------------------------------------------------------
__global__ void wmma_gemm_f16(const _Float16* __restrict__ A,
                              const _Float16* __restrict__ Bt,
                              const float* __restrict__ bias,
                              float* __restrict__ C,
                              int M, int N, int K) {
  const int lane = threadIdx.x & 31;
  const int wave = (blockIdx.x * blockDim.x + threadIdx.x) >> 5;
  const int tilesN = N >> 5;
  const int totalTiles = (M >> 5) * tilesN;
  if (wave >= totalTiles) return;            // whole-wave uniform: EXEC stays all-1s
  const int mt = wave / tilesN;
  const int nt = wave - mt * tilesN;
  const int h = lane >> 4;                   // half-wave select
  const int r = lane & 15;                   // A row / B col within 16-tile
  const _Float16* arow0 = A  + (size_t)(mt * 32 + r) * K;
  const _Float16* arow1 = arow0 + (size_t)16 * K;
  const _Float16* brow0 = Bt + (size_t)(nt * 32 + r) * K;
  const _Float16* brow1 = brow0 + (size_t)16 * K;
  v8f acc00 = {}, acc01 = {}, acc10 = {}, acc11 = {};
  for (int k0 = 0; k0 < K; k0 += 32) {
    Frag a0, a1, b0, b1;
    a0.h2[0] = *(const v8h*)(arow0 + k0 + h * 8);        // K = k0 + h*8 + 0..7
    a0.h2[1] = *(const v8h*)(arow0 + k0 + 16 + h * 8);   // K = k0 + 16 + h*8 + 0..7
    a1.h2[0] = *(const v8h*)(arow1 + k0 + h * 8);
    a1.h2[1] = *(const v8h*)(arow1 + k0 + 16 + h * 8);
    b0.h2[0] = *(const v8h*)(brow0 + k0 + h * 8);        // K = k0 + h*8 + 0..7
    b0.h2[1] = *(const v8h*)(brow0 + k0 + 16 + h * 8);   // K = k0 + 16 + h*8 + 0..7
    b1.h2[0] = *(const v8h*)(brow1 + k0 + h * 8);
    b1.h2[1] = *(const v8h*)(brow1 + k0 + 16 + h * 8);
    acc00 = __builtin_amdgcn_wmma_f32_16x16x32_f16(false, a0.v, false, b0.v,
                                                   (short)0, acc00, false, false);
    acc01 = __builtin_amdgcn_wmma_f32_16x16x32_f16(false, a0.v, false, b1.v,
                                                   (short)0, acc01, false, false);
    acc10 = __builtin_amdgcn_wmma_f32_16x16x32_f16(false, a1.v, false, b0.v,
                                                   (short)0, acc10, false, false);
    acc11 = __builtin_amdgcn_wmma_f32_16x16x32_f16(false, a1.v, false, b1.v,
                                                   (short)0, acc11, false, false);
  }
  const int col0 = nt * 32 + r;
  const int col1 = col0 + 16;
  const float bv0 = bias ? bias[col0] : 0.0f;
  const float bv1 = bias ? bias[col1] : 0.0f;
  float* c0 = C + (size_t)(mt * 32 + h * 8) * N;         // rows of upper 16-tile
  float* c1 = C + (size_t)(mt * 32 + 16 + h * 8) * N;    // rows of lower 16-tile
#pragma unroll
  for (int rr = 0; rr < 8; ++rr) {
    c0[(size_t)rr * N + col0] = acc00[rr] + bv0;
    c0[(size_t)rr * N + col1] = acc01[rr] + bv1;
    c1[(size_t)rr * N + col0] = acc10[rr] + bv0;
    c1[(size_t)rr * N + col1] = acc11[rr] + bv1;
  }
}

// ---------------------------------------------------------------------------
// Grouped MoE expert GEMM via WMMA. grid.x = expert, 512 threads = 16 waves,
// one wave per 16-col output tile. The wave's expert-weight B fragments
// (K=256 -> 8 frags) are preloaded into registers once and reused across all
// of that expert's gathered 16-token row chunks. out must be zeroed; each
// token gets exactly 2 atomic adds (commutative -> deterministic).
// ---------------------------------------------------------------------------
__global__ void moe_expert_gemm(const _Float16* __restrict__ A,   // (NTOK, 256) f16
                                const _Float16* __restrict__ WT,  // (NE, 256, 256) f16, (u,t)
                                const float* __restrict__ bias,   // (NE, 256)
                                const int* __restrict__ idx,      // token list, grouped
                                const int* __restrict__ offs,     // NE+1 offsets
                                float* __restrict__ out) {        // (NTOK, 256) f32
  const int e = blockIdx.x;
  const int lane = threadIdx.x & 31;
  const int nt = threadIdx.x >> 5;   // 0..15 column tile
  const int h = lane >> 4;
  const int r = lane & 15;
  const int begin = offs[e], end = offs[e + 1];
  if (begin == end) return;
  const _Float16* brow = WT + ((size_t)e * 256 + nt * 16 + r) * 256;
  Frag bf[8];
#pragma unroll
  for (int kk = 0; kk < 8; ++kk) {             // hoisted: B depends only on (e, nt)
    bf[kk].h2[0] = *(const v8h*)(brow + kk * 32 + h * 8);
    bf[kk].h2[1] = *(const v8h*)(brow + kk * 32 + 16 + h * 8);
  }
  const int col = nt * 16 + r;
  const float bv = bias[(size_t)e * 256 + col];
  for (int m0 = begin; m0 < end; m0 += 16) {
    int arowi = m0 + r;
    int tok = (arowi < end) ? idx[arowi] : idx[begin];   // safe pad row
    const _Float16* arow = A + (size_t)tok * 256;
    v8f acc = {};
#pragma unroll
    for (int kk = 0; kk < 8; ++kk) {
      Frag a;
      a.h2[0] = *(const v8h*)(arow + kk * 32 + h * 8);
      a.h2[1] = *(const v8h*)(arow + kk * 32 + 16 + h * 8);
      acc = __builtin_amdgcn_wmma_f32_16x16x32_f16(false, a.v, false, bf[kk].v,
                                                   (short)0, acc, false, false);
    }
#pragma unroll
    for (int rr = 0; rr < 8; ++rr) {
      int orow = m0 + h * 8 + rr;
      if (orow < end) {
        int otok = idx[orow];
        atomicAdd(&out[(size_t)otok * 256 + col], acc[rr] + bv);
      }
    }
  }
}

// ------------------------------ prep kernels -------------------------------
// GCN: W0+W1 transposed to f16 (N=256, K=512); bias = -(b0+b1).
__global__ void prep_gcn(const float* __restrict__ w, const float* __restrict__ bvec,
                         _Float16* __restrict__ wt, float* __restrict__ bneg) {
  int i = blockIdx.x * blockDim.x + threadIdx.x;
  if (i >= TIN * TT) return;
  int k = i / TT, n = i - k * TT;
  wt[(size_t)n * TIN + k] = (_Float16)(w[i] + w[TIN * TT + i]);
  if (k == 0) bneg[n] = -(bvec[n] + bvec[TT + n]);
}

// Batched transpose-to-f16: nmat matrices of (K,N) -> (N,K) f16.
__global__ void transpose_to_f16(const float* __restrict__ w, _Float16* __restrict__ wt,
                                 int K, int N, int nmat) {
  long i = (long)blockIdx.x * blockDim.x + threadIdx.x;
  long total = (long)nmat * K * N;
  if (i >= total) return;
  long per = (long)K * N;
  int m = (int)(i / per);
  long rem = i - (long)m * per;
  int k = (int)(rem / N), n = (int)(rem - (long)k * N);
  wt[(size_t)m * per + (size_t)n * K + k] = (_Float16)w[i];
}

__global__ void norm_centers(const float* __restrict__ c, float* __restrict__ cn, int rows) {
  int r = blockIdx.x * blockDim.x + threadIdx.x;
  if (r >= rows) return;
  float s = 0.f;
  for (int d = 0; d < NED; ++d) { float v = c[r * NED + d]; s += v * v; }
  float inv = 1.f / fmaxf(sqrtf(s), 1e-12f);
  for (int d = 0; d < NED; ++d) cn[r * NED + d] = c[r * NED + d] * inv;
}

// --------------------------- elementwise kernels ---------------------------
__global__ void cvt_f16(const float* __restrict__ x, _Float16* __restrict__ y, int n) {
  int i = blockIdx.x * blockDim.x + threadIdx.x;
  if (i < n) y[i] = (_Float16)x[i];
}
__global__ void diff_f16(const float* __restrict__ a, const float* __restrict__ b,
                         _Float16* __restrict__ y, int n) {
  int i = blockIdx.x * blockDim.x + threadIdx.x;
  if (i < n) y[i] = (_Float16)(a[i] - b[i]);
}
__global__ void concat2_f16(const float* __restrict__ a, const float* __restrict__ b,
                            _Float16* __restrict__ y, int rows) {
  int i = blockIdx.x * blockDim.x + threadIdx.x;
  if (i >= rows * 2 * TT) return;
  int r = i / (2 * TT), j = i - r * (2 * TT);
  y[i] = (_Float16)(j < TT ? a[r * TT + j] : b[r * TT + (j - TT)]);
}
__global__ void build_h16(const float* __restrict__ gl, const float* __restrict__ gr,
                          _Float16* __restrict__ h) {
  int i = blockIdx.x * blockDim.x + threadIdx.x;
  const int PER = NC2 * TT;        // 8960
  if (i >= NB * 2 * PER) return;
  int b = i / (2 * PER), j = i - b * (2 * PER);
  float v = (j < PER) ? gl[(size_t)b * PER + j] : gr[(size_t)b * PER + (j - PER)];
  h[i] = (_Float16)v;
}
__global__ void elu_to_f16(const float* __restrict__ x, _Float16* __restrict__ y, int n) {
  int i = blockIdx.x * blockDim.x + threadIdx.x;
  if (i >= n) return;
  float v = x[i];
  y[i] = (_Float16)(v > 0.f ? v : (expf(v) - 1.f));
}
__global__ void elu_f32(const float* __restrict__ x, float* __restrict__ y, int n) {
  int i = blockIdx.x * blockDim.x + threadIdx.x;
  if (i >= n) return;
  float v = x[i];
  y[i] = v > 0.f ? v : (expf(v) - 1.f);
}

// LayerNorm over last dim 256 (1 block per row). Optional relu pre, residual post.
__global__ void ln256(const float* __restrict__ x, const float* __restrict__ gamma,
                      const float* __restrict__ beta, const float* __restrict__ res,
                      int do_relu, float* __restrict__ out) {
  int row = blockIdx.x, t = threadIdx.x;
  float v = x[(size_t)row * TT + t];
  if (do_relu) v = fmaxf(v, 0.f);
  __shared__ float red[TT];
  red[t] = v; __syncthreads();
  for (int s = 128; s > 0; s >>= 1) { if (t < s) red[t] += red[t + s]; __syncthreads(); }
  float mu = red[0] * (1.f / TT);
  __syncthreads();
  float d = v - mu;
  red[t] = d * d; __syncthreads();
  for (int s = 128; s > 0; s >>= 1) { if (t < s) red[t] += red[t + s]; __syncthreads(); }
  float var = red[0] * (1.f / TT);
  float y = d * rsqrtf(var + 1e-5f) * gamma[t] + beta[t];
  if (res) y += res[(size_t)row * TT + t];
  out[(size_t)row * TT + t] = y;
}

// ------------------------------ attention ----------------------------------
// scores + softmax over 35 keys; 1 block per (b, q-row).
__global__ void attn_scores(const float* __restrict__ q, const float* __restrict__ k,
                            float* __restrict__ sc) {
  int b = blockIdx.x / NC2, qi = blockIdx.x - b * NC2;
  int t = threadIdx.x;
  __shared__ float srow[NC2];
  if (t < NC2) {
    const float* qp = q + (size_t)(b * NC2 + qi) * TT;
    const float* kp = k + (size_t)(b * NC2 + t) * TT;
    float s = 0.f;
    for (int d = 0; d < TT; ++d) s += qp[d] * kp[d];
    srow[t] = s * (1.f / 16.f);       // /sqrt(256)
  }
  __syncthreads();
  if (t == 0) {
    float mx = -1e30f;
    for (int j = 0; j < NC2; ++j) mx = fmaxf(mx, srow[j]);
    float sum = 0.f;
    for (int j = 0; j < NC2; ++j) { srow[j] = expf(srow[j] - mx); sum += srow[j]; }
    float inv = 1.f / sum;
    for (int j = 0; j < NC2; ++j) srow[j] *= inv;
  }
  __syncthreads();
  if (t < NC2) sc[(size_t)(b * NC2 + qi) * NC2 + t] = srow[t];
}

// o_l = sc @ v, o_r = sc^T @ v; 1 block per (b, q-row), 256 threads.
__global__ void attn_apply(const float* __restrict__ sc, const float* __restrict__ v,
                           float* __restrict__ ol, float* __restrict__ orr) {
  int b = blockIdx.x / NC2, qi = blockIdx.x - b * NC2;
  int d = threadIdx.x;
  __shared__ float srow[NC2], scol[NC2];
  if (d < NC2) {
    srow[d] = sc[(size_t)(b * NC2 + qi) * NC2 + d];
    scol[d] = sc[(size_t)(b * NC2 + d) * NC2 + qi];
  }
  __syncthreads();
  float al = 0.f, ar = 0.f;
  for (int kk = 0; kk < NC2; ++kk) {
    float vv = v[(size_t)(b * NC2 + kk) * TT + d];
    al += srow[kk] * vv;
    ar += scol[kk] * vv;
  }
  ol[(size_t)(b * NC2 + qi) * TT + d] = al;
  orr[(size_t)(b * NC2 + qi) * TT + d] = ar;
}

// ------------------------------ MoE routing --------------------------------
// l2norm(z) -> cosine sim vs normalized centers -> top-2 (P = softmax-sum = 1).
__global__ void router_topk(const float* __restrict__ z, const float* __restrict__ cn,
                            int* __restrict__ e1, int* __restrict__ e2, int ntok) {
  int tok = blockIdx.x * blockDim.x + threadIdx.x;
  if (tok >= ntok) return;
  float zv[NED]; float nrm = 0.f;
  for (int d = 0; d < NED; ++d) { zv[d] = z[(size_t)tok * NED + d]; nrm += zv[d] * zv[d]; }
  float inv = 1.f / fmaxf(sqrtf(nrm), 1e-12f);
  float b1 = -1e30f, b2 = -1e30f; int i1 = 0, i2 = 0;
  for (int e = 0; e < NE; ++e) {
    float s = 0.f;
    for (int d = 0; d < NED; ++d) s += zv[d] * cn[e * NED + d];
    s *= inv;
    if (s > b1) { b2 = b1; i2 = i1; b1 = s; i1 = e; }
    else if (s > b2) { b2 = s; i2 = e; }
  }
  e1[tok] = i1; e2[tok] = i2;
}

// Deterministic per-expert token lists (serial per-expert scan).
__global__ void moe_count(const int* __restrict__ e1, const int* __restrict__ e2,
                          int* __restrict__ cnt, int ntok) {
  int e = threadIdx.x;
  if (e >= NE) return;
  int c = 0;
  for (int t = 0; t < ntok; ++t) c += (e1[t] == e) + (e2[t] == e);
  cnt[e] = c;
}
__global__ void moe_scan(const int* __restrict__ cnt, int* __restrict__ offs) {
  if (threadIdx.x == 0) {
    int s = 0;
    for (int e = 0; e < NE; ++e) { offs[e] = s; s += cnt[e]; }
    offs[NE] = s;
  }
}
__global__ void moe_fill(const int* __restrict__ e1, const int* __restrict__ e2,
                         const int* __restrict__ offs, int* __restrict__ idx, int ntok) {
  int e = threadIdx.x;
  if (e >= NE) return;
  int p = offs[e];
  for (int t = 0; t < ntok; ++t) {
    if (e1[t] == e) idx[p++] = t;
    if (e2[t] == e) idx[p++] = t;
  }
}

// ------------------------------ classifier tail ----------------------------
__global__ void fc3_kernel(const float* __restrict__ h2, const float* __restrict__ w,
                           const float* __restrict__ b, float* __restrict__ out) {
  int i = blockIdx.x * blockDim.x + threadIdx.x;
  if (i >= NB * NCL) return;
  int bb = i / NCL, n = i - bb * NCL;
  float s = b[n];
  for (int kk = 0; kk < 32; ++kk) s += h2[bb * 32 + kk] * w[kk * NCL + n];
  out[i] = s;
}

// ===========================================================================
extern "C" void kernel_launch(void* const* d_in, const int* in_sizes, int n_in,
                              void* d_out, int out_size, void* d_ws, size_t ws_size,
                              hipStream_t stream) {
  (void)in_sizes; (void)n_in; (void)out_size; (void)ws_size;
  const float* x_l        = (const float*)d_in[0];
  const float* x_r        = (const float*)d_in[1];
  const float* gcn_w      = (const float*)d_in[2];
  const float* gcn_b      = (const float*)d_in[3];
  const float* bidgn_ln_g = (const float*)d_in[4];
  const float* bidgn_ln_b = (const float*)d_in[5];
  const float* attn_w     = (const float*)d_in[6];
  const float* attn_b     = (const float*)d_in[7];
  const float* attn_ln_g  = (const float*)d_in[8];
  const float* attn_ln_b  = (const float*)d_in[9];
  const float* router_w   = (const float*)d_in[10];
  const float* router_b   = (const float*)d_in[11];
  const float* centers    = (const float*)d_in[12];
  const float* expert_w   = (const float*)d_in[13];
  const float* expert_b   = (const float*)d_in[14];
  const float* moe_ln_g   = (const float*)d_in[15];
  const float* moe_ln_b   = (const float*)d_in[16];
  const float* fc1_w      = (const float*)d_in[17];
  const float* fc1_b      = (const float*)d_in[18];
  const float* fc2_w      = (const float*)d_in[19];
  const float* fc2_b      = (const float*)d_in[20];
  const float* fc3_w      = (const float*)d_in[21];
  const float* fc3_b      = (const float*)d_in[22];

  // -------- workspace layout --------
  size_t off = 0;
  auto alloc = [&](size_t bytes) -> void* {
    void* p = (char*)d_ws + off;
    off += (bytes + 255) & ~(size_t)255;
    return p;
  };
  _Float16* wgcn_t    = (_Float16*)alloc((size_t)TT * TIN * 2);
  float*    gcn_bneg  = (float*)   alloc(TT * 4);
  _Float16* wattn_t   = (_Float16*)alloc((size_t)8 * TT * TT * 2);
  _Float16* wrouter_t = (_Float16*)alloc((size_t)2 * NED * TIN * 2);
  float*    cn        = (float*)   alloc((size_t)2 * NE * NED * 4);
  _Float16* wfc1_t    = (_Float16*)alloc((size_t)TT * (2 * NC2 * TT) * 2);
  _Float16* wfc2_t    = (_Float16*)alloc((size_t)32 * TT * 2);
  _Float16* wexp_t    = (_Float16*)alloc((size_t)NE * TT * TT * 2);
  _Float16* a16       = (_Float16*)alloc((size_t)NTOK * TIN * 2);     // x / xc
  _Float16* b16a      = (_Float16*)alloc((size_t)NTOK * TT * 2);
  _Float16* b16b      = (_Float16*)alloc((size_t)NTOK * TT * 2);
  _Float16* b16c      = (_Float16*)alloc((size_t)NTOK * TT * 2);
  _Float16* h16       = (_Float16*)alloc((size_t)NB * 2 * NC2 * TT * 2);
  _Float16* h1_16     = (_Float16*)alloc((size_t)NB * TT * 2);
  float* g_l  = (float*)alloc((size_t)NTOK * TT * 4);
  float* g_r  = (float*)alloc((size_t)NTOK * TT * 4);
  float* qb   = (float*)alloc((size_t)NTOK * TT * 4);
  float* kb   = (float*)alloc((size_t)NTOK * TT * 4);
  float* vb   = (float*)alloc((size_t)NTOK * TT * 4);
  float* ol   = (float*)alloc((size_t)NTOK * TT * 4);
  float* orr  = (float*)alloc((size_t)NTOK * TT * 4);
  float* a_l  = (float*)alloc((size_t)NTOK * TT * 4);
  float* a_r  = (float*)alloc((size_t)NTOK * TT * 4);
  float* tmp1 = (float*)alloc((size_t)NTOK * TT * 4);
  float* sc   = (float*)alloc((size_t)NB * NC2 * NC2 * 4);
  float* zb   = (float*)alloc((size_t)NTOK * NED * 4);
  int*   e1   = (int*)alloc(NTOK * 4);
  int*   e2   = (int*)alloc(NTOK * 4);
  int*   cnt  = (int*)alloc(NE * 4);
  int*   offs = (int*)alloc((NE + 1) * 4);
  int*   idxl = (int*)alloc(NTOK * 2 * 4);
  float* h1   = (float*)alloc((size_t)NB * TT * 4);
  float* h2   = (float*)alloc((size_t)NB * 32 * 4);
  float* h2e  = (float*)alloc((size_t)NB * 32 * 4);

  auto grid = [](long n) { return dim3((unsigned)((n + 255) / 256)); };
  auto gemm = [&](const _Float16* A, const _Float16* Bt, const float* bias,
                  float* C, int M, int N, int K) {
    int tiles = (M / 32) * (N / 32);
    int blocks = (tiles + 7) / 8;
    wmma_gemm_f16<<<blocks, 256, 0, stream>>>(A, Bt, bias, C, M, N, K);
  };

  // -------- weight prep (f16 transposed) --------
  prep_gcn<<<grid((long)TIN * TT), 256, 0, stream>>>(gcn_w, gcn_b, wgcn_t, gcn_bneg);
  transpose_to_f16<<<grid((long)8 * TT * TT), 256, 0, stream>>>(attn_w, wattn_t, TT, TT, 8);
  transpose_to_f16<<<grid((long)2 * TIN * NED), 256, 0, stream>>>(router_w, wrouter_t, TIN, NED, 2);
  transpose_to_f16<<<grid((long)(2 * NC2 * TT) * TT), 256, 0, stream>>>(fc1_w, wfc1_t, 2 * NC2 * TT, TT, 1);
  transpose_to_f16<<<grid((long)TT * 32), 256, 0, stream>>>(fc2_w, wfc2_t, TT, 32, 1);
  norm_centers<<<grid(2 * NE), 256, 0, stream>>>(centers, cn, 2 * NE);

  // -------- BiDGN (Chebynet K=2 with identity supports) --------
  cvt_f16<<<grid((long)NTOK * TIN), 256, 0, stream>>>(x_l, a16, NTOK * TIN);
  gemm(a16, wgcn_t, gcn_bneg, tmp1, NTOK, TT, TIN);
  ln256<<<NTOK, TT, 0, stream>>>(tmp1, bidgn_ln_g, bidgn_ln_b, nullptr, 1, g_l);
  cvt_f16<<<grid((long)NTOK * TIN), 256, 0, stream>>>(x_r, a16, NTOK * TIN);
  gemm(a16, wgcn_t, gcn_bneg, tmp1, NTOK, TT, TIN);
  ln256<<<NTOK, TT, 0, stream>>>(tmp1, bidgn_ln_g, bidgn_ln_b, nullptr, 1, g_r);

  // -------- two BiDGNBlocks --------
  for (int i = 0; i < 2; ++i) {
    const _Float16* WqT = wattn_t + (size_t)(i * 4 + 0) * TT * TT;
    const _Float16* WkT = wattn_t + (size_t)(i * 4 + 1) * TT * TT;
    const _Float16* WvT = wattn_t + (size_t)(i * 4 + 2) * TT * TT;
    const _Float16* WpT = wattn_t + (size_t)(i * 4 + 3) * TT * TT;

    // q/k/v
    cvt_f16<<<grid((long)NTOK * TT), 256, 0, stream>>>(g_l, b16a, NTOK * TT);
    cvt_f16<<<grid((long)NTOK * TT), 256, 0, stream>>>(g_r, b16b, NTOK * TT);
    diff_f16<<<grid((long)NTOK * TT), 256, 0, stream>>>(g_l, g_r, b16c, NTOK * TT);
    gemm(b16a, WqT, attn_b + (i * 4 + 0) * TT, qb, NTOK, TT, TT);
    gemm(b16b, WkT, attn_b + (i * 4 + 1) * TT, kb, NTOK, TT, TT);
    gemm(b16c, WvT, attn_b + (i * 4 + 2) * TT, vb, NTOK, TT, TT);

    // attention (35-wide, VALU)
    attn_scores<<<NB * NC2, 64, 0, stream>>>(qb, kb, sc);
    attn_apply<<<NB * NC2, TT, 0, stream>>>(sc, vb, ol, orr);

    // projection + LN + residual
    cvt_f16<<<grid((long)NTOK * TT), 256, 0, stream>>>(ol, b16a, NTOK * TT);
    cvt_f16<<<grid((long)NTOK * TT), 256, 0, stream>>>(orr, b16b, NTOK * TT);
    gemm(b16a, WpT, attn_b + (i * 4 + 3) * TT, tmp1, NTOK, TT, TT);
    ln256<<<NTOK, TT, 0, stream>>>(tmp1, attn_ln_g + (i * 2 + 0) * TT,
                                   attn_ln_b + (i * 2 + 0) * TT, g_l, 0, a_l);
    gemm(b16b, WpT, attn_b + (i * 4 + 3) * TT, tmp1, NTOK, TT, TT);
    ln256<<<NTOK, TT, 0, stream>>>(tmp1, attn_ln_g + (i * 2 + 1) * TT,
                                   attn_ln_b + (i * 2 + 1) * TT, g_r, 0, a_r);

    // router: z = xc @ W + b, then cosine top-2 (P == 1)
    concat2_f16<<<grid((long)NTOK * 2 * TT), 256, 0, stream>>>(a_l, a_r, a16, NTOK);
    gemm(a16, wrouter_t + (size_t)i * NED * TIN, router_b + i * NED, zb, NTOK, NED, 2 * TT);
    router_topk<<<grid(NTOK), 256, 0, stream>>>(zb, cn + (size_t)i * NE * NED, e1, e2, NTOK);
    moe_count<<<1, 96, 0, stream>>>(e1, e2, cnt, NTOK);
    moe_scan<<<1, 32, 0, stream>>>(cnt, offs);
    moe_fill<<<1, 96, 0, stream>>>(e1, e2, offs, idxl, NTOK);

    // expert weights for this block -> f16 transposed
    transpose_to_f16<<<grid((long)NE * TT * TT), 256, 0, stream>>>(
        expert_w + (size_t)i * NE * TT * TT, wexp_t, TT, TT, NE);

    // grouped expert GEMM + LN + residual (both sides share routing)
    cvt_f16<<<grid((long)NTOK * TT), 256, 0, stream>>>(a_l, b16a, NTOK * TT);
    cvt_f16<<<grid((long)NTOK * TT), 256, 0, stream>>>(a_r, b16b, NTOK * TT);
    hipMemsetAsync(tmp1, 0, (size_t)NTOK * TT * 4, stream);
    moe_expert_gemm<<<NE, 512, 0, stream>>>(b16a, wexp_t, expert_b + (size_t)i * NE * TT,
                                            idxl, offs, tmp1);
    ln256<<<NTOK, TT, 0, stream>>>(tmp1, moe_ln_g + (i * 2 + 0) * TT,
                                   moe_ln_b + (i * 2 + 0) * TT, a_l, 0, g_l);
    hipMemsetAsync(tmp1, 0, (size_t)NTOK * TT * 4, stream);
    moe_expert_gemm<<<NE, 512, 0, stream>>>(b16b, wexp_t, expert_b + (size_t)i * NE * TT,
                                            idxl, offs, tmp1);
    ln256<<<NTOK, TT, 0, stream>>>(tmp1, moe_ln_g + (i * 2 + 1) * TT,
                                   moe_ln_b + (i * 2 + 1) * TT, a_r, 0, g_r);
  }

  // -------- classifier head --------
  build_h16<<<grid((long)NB * 2 * NC2 * TT), 256, 0, stream>>>(g_l, g_r, h16);
  gemm(h16, wfc1_t, fc1_b, h1, NB, TT, 2 * NC2 * TT);
  elu_to_f16<<<grid((long)NB * TT), 256, 0, stream>>>(h1, h1_16, NB * TT);
  gemm(h1_16, wfc2_t, fc2_b, h2, NB, 32, TT);
  elu_f32<<<grid(NB * 32), 256, 0, stream>>>(h2, h2e, NB * 32);
  fc3_kernel<<<1, 256, 0, stream>>>(h2e, fc3_w, fc3_b, (float*)d_out);
}
